// CWLNFace_11201274708639
// MI455X (gfx1250) — matrix operand order repaired and checked by reading, not compiled, not run on hardware
//
#include <hip/hip_runtime.h>
#include <math.h>

#define B_SZ   512
#define EMB_D  512
#define CLS    70722

#define BLK_M  128
#define BLK_N  128
#define KSTEP  32
#define PAD    40      // bf16 elements per LDS row (32 data + 8 pad -> 80B stride, conflict-friendly)

#define MARG   0.4f
#define HC     0.333f
#define SSCALE 64.0f
#define EPSV   0.001f
#define PI_F   3.14159265358979323846f

typedef __attribute__((ext_vector_type(16))) __bf16 v16bf;
typedef __attribute__((ext_vector_type(8)))  __bf16 v8bf;
typedef __attribute__((ext_vector_type(8)))  float  v8f;
typedef __attribute__((ext_vector_type(4)))  float  v4f;

// ---------------------------------------------------------------------------
// Kernel 1: per-row margin scaler (mean/std over B with ddof=1).  512 threads.
// ---------------------------------------------------------------------------
__global__ __launch_bounds__(512) void cwln_stats(const float* __restrict__ norms,
                                                  const float* __restrict__ csn,
                                                  float* __restrict__ ms_out) {
  __shared__ float red[512];
  const int i = threadIdx.x;

  float sn = norms[i];
  sn = fminf(fmaxf(sn, 0.001f), 100.0f);
  sn = sn / (csn[i] + 0.001f);
  sn = fminf(fmaxf(sn, 0.001f), 100.0f);

  red[i] = sn;
  __syncthreads();
  for (int s = 256; s > 0; s >>= 1) {
    if (i < s) red[i] += red[i + s];
    __syncthreads();
  }
  const float mean = red[0] * (1.0f / 512.0f);
  __syncthreads();

  const float d = sn - mean;
  red[i] = d * d;
  __syncthreads();
  for (int s = 256; s > 0; s >>= 1) {
    if (i < s) red[i] += red[i + s];
    __syncthreads();
  }
  const float stdv = sqrtf(red[0] * (1.0f / 511.0f));   // ddof = 1

  float msc = (sn - mean) / (stdv + EPSV) * HC;
  msc = fminf(fmaxf(msc, -1.0f), 1.0f);
  ms_out[i] = msc;
}

// ---------------------------------------------------------------------------
// Fragment gather from LDS: two 16B ds loads -> one v16bf operand.
// ---------------------------------------------------------------------------
__device__ __forceinline__ v16bf ld_frag(const __bf16* p, int offElems) {
  v8bf x = *(const v8bf*)p;
  v8bf y = *(const v8bf*)(p + offElems);
  return __builtin_shufflevector(x, y, 0, 1, 2, 3, 4, 5, 6, 7,
                                       8, 9, 10, 11, 12, 13, 14, 15);
}

// ---------------------------------------------------------------------------
// Kernel 2: fused  [column-norm + GEMM(bf16x3 split) + clip/scale epilogue]
//   block tile 128M x 128N, 256 threads = 8 waves (2 in M x 4 in N),
//   wave tile 64M x 32N = 4x2 WMMA accumulators, K-step 32.
//
// Grid: x = M tile (4, innermost dispatch), y = N tile (553) so the 4 blocks
// sharing a 256KB kernel-matrix panel are dispatch-adjacent -> panel read
// from HBM once, served from L2 (192MB) afterwards.
//
// Epilogue is branch-free: cos(clip(acos(clip(c)),eps,pi-eps)) == clip(c) for
// every non-label element (acos of +-(1-1e-3) lies in [0.0447, pi-0.0447], so
// the theta clip never binds).  The 512 label elements are patched by a
// separate tiny kernel -> no transcendentals / exec-mask churn in this kernel.
// ---------------------------------------------------------------------------
__global__ __launch_bounds__(256) void cwln_gemm(const float* __restrict__ emb,
                                                 const float* __restrict__ kw,
                                                 float* __restrict__ out) {
  __shared__ __attribute__((aligned(16))) __bf16 ldsAhi[BLK_M * PAD];
  __shared__ __attribute__((aligned(16))) __bf16 ldsAlo[BLK_M * PAD];
  __shared__ __attribute__((aligned(16))) __bf16 ldsBhi[BLK_N * PAD];
  __shared__ __attribute__((aligned(16))) __bf16 ldsBlo[BLK_N * PAD];
  __shared__ float ldsSq[8 * BLK_N];
  __shared__ float ldsInv[BLK_N];

  const int t     = threadIdx.x;
  const int mBase = blockIdx.x * BLK_M;   // 4 M tiles, innermost for L2 reuse
  const int nBase = blockIdx.y * BLK_N;   // 553 N tiles

  const int w     = t >> 5;
  const int lane  = t & 31;
  const int wm    = w & 1;                // 2 waves along M
  const int wn    = w >> 1;               // 4 waves along N
  const int halfK = lane >> 4;
  const int l15   = lane & 15;

  v8f acc[4][2] = {};                     // 4 M-frags x 2 N-frags, f32 16x16 each

  // loader mapping (256 threads)
  const int aRow0 = t >> 3;               // + i*32 : A rows
  const int aCol  = (t & 7) * 4;          // float4 column within K-chunk
  const int bK0   = t >> 5;               // + i*8  : B k-rows
  const int bCol  = t & 31;               // + d*32 : B columns (coalesced b32 loads)

  float sq[4] = {0.f, 0.f, 0.f, 0.f};     // per-thread column sum-of-squares

  for (int k0 = 0; k0 < EMB_D; k0 += KSTEP) {
    // ---- stage A chunk (128 x 32 fp32) -> bf16 hi/lo planes ----
#pragma unroll
    for (int i = 0; i < 4; ++i) {
      const int row = aRow0 + i * 32;
      v4f a = *(const v4f*)(emb + (size_t)(mBase + row) * EMB_D + k0 + aCol);
#pragma unroll
      for (int d = 0; d < 4; ++d) {
        const float x = a[d];
        const __bf16 h = (__bf16)x;
        const __bf16 l = (__bf16)(x - (float)h);
        ldsAhi[row * PAD + aCol + d] = h;
        ldsAlo[row * PAD + aCol + d] = l;
      }
    }

    // ---- stage B chunk (32 x 128 fp32) transposed -> bf16 hi/lo + sumsq ----
#pragma unroll
    for (int i = 0; i < 4; ++i) {
      const int kr = bK0 + i * 8;
      const float* src = kw + (size_t)(k0 + kr) * CLS + nBase;
      if (k0 + KSTEP < EMB_D)
        __builtin_prefetch(src + (size_t)KSTEP * CLS, 0, 1);  // next K-chunk
#pragma unroll
      for (int d = 0; d < 4; ++d) {
        const int c = bCol + d * 32;
        float x = 0.0f;
        if (nBase + c < CLS) x = __builtin_nontemporal_load(src + c);
        sq[d] += x * x;
        const __bf16 h = (__bf16)x;
        const __bf16 l = (__bf16)(x - (float)h);
        ldsBhi[c * PAD + kr] = h;
        ldsBlo[c * PAD + kr] = l;
      }
    }
    __syncthreads();

    // ---- B fragments for this wave's two N tiles ----
    v16bf bhi[2], blo[2];
#pragma unroll
    for (int n = 0; n < 2; ++n) {
      const int c = wn * 32 + n * 16 + l15;
      bhi[n] = ld_frag(&ldsBhi[c * PAD + halfK * 8], 16);
      blo[n] = ld_frag(&ldsBlo[c * PAD + halfK * 8], 16);
    }

    // ---- A fragments + 24 WMMAs (bf16x3 split accumulation) ----
#pragma unroll
    for (int f = 0; f < 4; ++f) {
      const int r = wm * 64 + f * 16 + l15;
      v16bf ahi = ld_frag(&ldsAhi[r * PAD + halfK * 8], 16);
      v16bf alo = ld_frag(&ldsAlo[r * PAD + halfK * 8], 16);
#pragma unroll
      for (int n = 0; n < 2; ++n) {
        acc[f][n] = __builtin_amdgcn_wmma_f32_16x16x32_bf16(
            false, ahi, false, bhi[n], (short)0, acc[f][n], false, false);
        acc[f][n] = __builtin_amdgcn_wmma_f32_16x16x32_bf16(
            false, alo, false, bhi[n], (short)0, acc[f][n], false, false);
        acc[f][n] = __builtin_amdgcn_wmma_f32_16x16x32_bf16(
            false, ahi, false, blo[n], (short)0, acc[f][n], false, false);
      }
    }
    __syncthreads();   // tail barrier: no conditional -> no loop peeling
  }

  // ---- fused column inverse-norm: reduce per-thread sumsq across 8 k-groups ----
  ldsSq[bK0 * BLK_N + bCol]      = sq[0];
  ldsSq[bK0 * BLK_N + bCol + 32] = sq[1];
  ldsSq[bK0 * BLK_N + bCol + 64] = sq[2];
  ldsSq[bK0 * BLK_N + bCol + 96] = sq[3];
  __syncthreads();
  if (t < BLK_N) {
    float s = 0.0f;
#pragma unroll
    for (int g = 0; g < 8; ++g) s += ldsSq[g * BLK_N + t];
    ldsInv[t] = rsqrtf(s);
  }
  __syncthreads();

  // ---- branch-free epilogue: scale by inv-norm, clip, scale, NT store ----
#pragma unroll
  for (int f = 0; f < 4; ++f) {
#pragma unroll
    for (int n = 0; n < 2; ++n) {
      const int cl  = wn * 32 + n * 16 + l15;
      const int col = nBase + cl;
      if (col >= CLS) continue;
      const float inv = ldsInv[cl];
#pragma unroll
      for (int v = 0; v < 8; ++v) {
        const int rl = wm * 64 + f * 16 + halfK * 8 + v;  // C layout: VGPR v -> M
        float c = acc[f][n][v] * inv;
        c = fminf(fmaxf(c, -1.0f + EPSV), 1.0f - EPSV);
        __builtin_nontemporal_store(c * SSCALE,
                                    out + (size_t)(mBase + rl) * CLS + col);
      }
    }
  }
}

// ---------------------------------------------------------------------------
// Kernel 3: patch the 512 label elements with the full margin path.
//   out[row,lab] currently holds clip(cosine)*S; S is a power of two so the
//   *(1/S) recovery is exact.
// ---------------------------------------------------------------------------
__global__ __launch_bounds__(512) void cwln_fixup(const int*   __restrict__ label,
                                                  const float* __restrict__ ms,
                                                  float* __restrict__ out) {
  const int i = threadIdx.x;
  const size_t idx = (size_t)i * CLS + label[i];
  const float c   = out[idx] * (1.0f / SSCALE);     // clipped cosine
  const float msr = ms[i];
  float th = acosf(c) - MARG * msr;
  th = fminf(fmaxf(th, EPSV), PI_F - EPSV);
  out[idx] = (cosf(th) - (MARG + MARG * msr)) * SSCALE;
}

// ---------------------------------------------------------------------------
extern "C" void kernel_launch(void* const* d_in, const int* in_sizes, int n_in,
                              void* d_out, int out_size, void* d_ws, size_t ws_size,
                              hipStream_t stream) {
  const float* emb   = (const float*)d_in[0];   // (512, 512)
  const float* norms = (const float*)d_in[1];   // (512, 1)
  const int*   label = (const int*)  d_in[2];   // (512,)
  const float* csn   = (const float*)d_in[3];   // (512,)
  const float* kw    = (const float*)d_in[4];   // (512, 70722)
  float* out = (float*)d_out;                   // (512, 70722)
  float* ms  = (float*)d_ws;                    // 512 floats scratch

  cwln_stats<<<1, 512, 0, stream>>>(norms, csn, ms);

  // x = M tiles (4, dispatch-adjacent for kernel-panel L2 reuse), y = N tiles
  dim3 grid(B_SZ / BLK_M, (CLS + BLK_N - 1) / BLK_N);   // 4 x 553
  cwln_gemm<<<grid, 256, 0, stream>>>(emb, kw, out);

  cwln_fixup<<<1, 512, 0, stream>>>(label, ms, out);
}
